// SelfAttention_12841952215534
// MI455X (gfx1250) — matrix operand order repaired and checked by reading, not compiled, not run on hardware
//
#include <hip/hip_runtime.h>
#include <hip/hip_bf16.h>

typedef __attribute__((ext_vector_type(16))) __bf16 v16bf;
typedef __attribute__((ext_vector_type(8)))  float  v8f;
typedef __attribute__((ext_vector_type(8)))  unsigned int v8u;
typedef unsigned int   u32;
typedef unsigned short u16;

#define S_LEN 2048
#define EMB   1024
#define HEADS 16
#define HD    64
#define BATCH 2

// fp32 -> bf16, round-to-nearest-even
__device__ __forceinline__ u16 f2bf(float f) {
  u32 u = __builtin_bit_cast(u32, f);
  u = (u + 0x7FFFu + ((u >> 16) & 1u)) >> 16;
  return (u16)u;
}

// Build a 16-element bf16 fragment from two 16-byte chunks (LDS or global).
__device__ __forceinline__ v16bf make_frag(const u16* c0, const u16* c1) {
  uint4 a = *(const uint4*)c0;
  uint4 b = *(const uint4*)c1;
  v8u v;
  v[0]=a.x; v[1]=a.y; v[2]=a.z; v[3]=a.w;
  v[4]=b.x; v[5]=b.y; v[6]=b.z; v[7]=b.w;
  return __builtin_bit_cast(v16bf, v);
}

// gfx1250 async copy global -> LDS, 16 bytes per lane. ASYNCcnt-tracked.
// lds_off: low 32 bits of the flat LDS-aperture address == wave-relative LDS
// byte offset (ISA 10.2).
__device__ __forceinline__ void async_copy_b128(u32 lds_off, const void* gaddr) {
  asm volatile("global_load_async_to_lds_b128 %0, %1, off"
               :: "v"(lds_off), "v"(gaddr) : "memory");
}
__device__ __forceinline__ void wait_async0() {
  asm volatile("s_wait_asynccnt 0x0" ::: "memory");
}

// ---------------------------------------------------------------------------
// Kernel 0: one-pass fp32 -> bf16 conversion (x and the three W matrices).
// Removes per-tile reconversion from the GEMM inner loop and halves L2 traffic.
// ---------------------------------------------------------------------------
__global__ __launch_bounds__(256) void f32_to_bf16_kernel(
    const float* __restrict__ in, u16* __restrict__ out, int n4)
{
  int i = blockIdx.x * 256 + threadIdx.x;
  if (i >= n4) return;
  float4 v = ((const float4*)in)[i];
  u32 lo = (u32)f2bf(v.x) | ((u32)f2bf(v.y) << 16);
  u32 hi = (u32)f2bf(v.z) | ((u32)f2bf(v.w) << 16);
  ((uint2*)out)[i] = make_uint2(lo, hi);
}

// ---------------------------------------------------------------------------
// Kernel 1: Y = X @ W^T  (M=4096, N=1024, K=1024), bf16 in, bf16 out.
// Double-buffered LDS fed by GLOBAL_LOAD_ASYNC_TO_LDS_B128.
// MODE 0: out[(b*H+h)][s][d]   (Q, K)
// MODE 1: out[(b*H+h)][d][s]   (V transposed for the P@V B-fragments)
// ---------------------------------------------------------------------------
#define LDS_STR 40  // halves per LDS row (32 + 8 pad -> conflict-free b128 reads)

// Stage one 128x32 bf16 tile pair (X rows @ mtile, W rows @ ntile) via async DMA.
__device__ __forceinline__ void stage_tiles_async(
    const u16* __restrict__ Xg, const u16* __restrict__ Wg,
    u16* xs, u16* ws, int t, int mtile, int ntile, int k0)
{
  #pragma unroll
  for (int i = 0; i < 2; ++i) {
    int c   = t + i * 256;        // chunk id 0..511
    int row = c >> 2;             // 0..127
    int kc  = (c & 3) * 8;        // half offset within row: 0,8,16,24
    u32 lofs = (u32)(row * LDS_STR + kc) * 2u;
    async_copy_b128((u32)(size_t)xs + lofs, Xg + (size_t)(mtile + row) * EMB + k0 + kc);
    async_copy_b128((u32)(size_t)ws + lofs, Wg + (size_t)(ntile + row) * EMB + k0 + kc);
  }
}

template<int MODE>
__global__ __launch_bounds__(256) void qkv_gemm_kernel(
    const u16* __restrict__ X, const u16* __restrict__ W,
    u16* __restrict__ Out)
{
  __shared__ alignas(16) u16 Xs[2][128 * LDS_STR];
  __shared__ alignas(16) u16 Ws[2][128 * LDS_STR];

  const int t    = threadIdx.x;
  const int lane = t & 31;
  const int w    = t >> 5;        // 0..7
  const int wm   = w & 1;         // wave M position (2)
  const int wn   = w >> 1;        // wave N position (4)
  const int h16  = lane >> 4;     // lane half
  const int ln   = lane & 15;

  const int mtile = blockIdx.y * 128;
  const int ntile = blockIdx.x * 128;

  v8f acc[4][2];
  #pragma unroll
  for (int i = 0; i < 4; ++i)
    #pragma unroll
    for (int j = 0; j < 2; ++j) acc[i][j] = {};

  // prologue: async-stage tile 0 into buffer 0
  stage_tiles_async(X, W, Xs[0], Ws[0], t, mtile, ntile, 0);

  for (int kk = 0; kk < EMB / 32; ++kk) {
    const int buf = kk & 1;
    wait_async0();       // my async copies (current tile) are done
    __syncthreads();     // => everyone's copies are done; prior reads retired
    if (kk + 1 < EMB / 32)
      stage_tiles_async(X, W, Xs[buf ^ 1], Ws[buf ^ 1], t, mtile, ntile, (kk + 1) * 32);

    const u16* xs = Xs[buf];
    const u16* ws = Ws[buf];

    v16bf bfrag[2];
    #pragma unroll
    for (int ni = 0; ni < 2; ++ni) {
      const u16* bp = &ws[(wn * 32 + ni * 16 + ln) * LDS_STR];
      bfrag[ni] = make_frag(bp + 8 * h16, bp + 16 + 8 * h16);
    }
    #pragma unroll
    for (int mi = 0; mi < 4; ++mi) {
      const u16* ap = &xs[(wm * 64 + mi * 16 + ln) * LDS_STR];
      v16bf afrag = make_frag(ap + 8 * h16, ap + 16 + 8 * h16);
      #pragma unroll
      for (int ni = 0; ni < 2; ++ni) {
        acc[mi][ni] = __builtin_amdgcn_wmma_f32_16x16x32_bf16(
            false, afrag, false, bfrag[ni], (short)0, acc[mi][ni], false, false);
      }
    }
  }

  // --- store bf16 in head-major layout (MODE folded at compile time) ---
  #pragma unroll
  for (int mi = 0; mi < 4; ++mi)
    #pragma unroll
    for (int ni = 0; ni < 2; ++ni)
      #pragma unroll
      for (int r = 0; r < 8; ++r) {
        int m = mtile + wm * 64 + mi * 16 + r + 8 * h16;
        int n = ntile + wn * 32 + ni * 16 + ln;
        int b = m >> 11, s = m & (S_LEN - 1);
        int h = n >> 6,  d = n & 63;
        size_t idx = (MODE == 0)
          ? ((size_t)(b * HEADS + h) * S_LEN + s) * HD + d
          : ((size_t)(b * HEADS + h) * HD + d) * S_LEN + s;
        Out[idx] = f2bf(acc[mi][ni][r]);
      }
}

// ---------------------------------------------------------------------------
// Kernel 2: flash-style attention with multiplicative tril mask.
// softmax over ( (q.k)*[k<=q] / 32 ); masked entries are 0, NOT -inf.
// grid: (S/128, B*H); 8 waves, 16 queries each, key blocks of 128.
// ---------------------------------------------------------------------------
#define P_STR 136   // 128 + 8 pad halves

__global__ __launch_bounds__(256) void attn_kernel(
    const u16* __restrict__ Qb, const u16* __restrict__ Kb,
    const u16* __restrict__ Vt, float* __restrict__ Out)
{
  __shared__ alignas(16) u16 Ps[8][16 * P_STR];  // per-wave P staging

  const int t    = threadIdx.x;
  const int lane = t & 31;
  const int w    = t >> 5;
  const int h16  = lane >> 4;
  const int ln   = lane & 15;

  const int bh = blockIdx.y;
  const int q0 = blockIdx.x * 128 + w * 16;

  const u16* Qh = Qb + (size_t)bh * S_LEN * HD;
  const u16* Kh = Kb + (size_t)bh * S_LEN * HD;
  const u16* Vh = Vt + (size_t)bh * HD * S_LEN;

  // Q A-fragments (2 K-steps over d=0..63), resident for whole kernel
  v16bf qfrag[2];
  {
    const u16* qrow = Qh + (size_t)(q0 + ln) * HD;
    #pragma unroll
    for (int ks = 0; ks < 2; ++ks) {
      const u16* p = qrow + ks * 32;
      qfrag[ks] = make_frag(p + 8 * h16, p + 16 + 8 * h16);
    }
  }

  float mrow[8], lrow[8];
  #pragma unroll
  for (int r = 0; r < 8; ++r) { mrow[r] = -1e30f; lrow[r] = 0.0f; }
  v8f oacc[4];
  #pragma unroll
  for (int i = 0; i < 4; ++i) oacc[i] = {};

  u16* pbuf = Ps[w];

  for (int kb = 0; kb < S_LEN; kb += 128) {
    // prefetch next key/value block into cache (global_prefetch_b8)
    if (kb + 128 < S_LEN) {
      __builtin_prefetch(Kh + (size_t)(kb + 128 + lane * 4) * HD, 0, 0);
      __builtin_prefetch(Vh + (size_t)(lane * 2) * S_LEN + kb + 128, 0, 0);
    }

    // ---- scores: 16x128, B-fragments straight from global (L2-resident) ----
    v8f sc[8];
    #pragma unroll
    for (int ntl = 0; ntl < 8; ++ntl) {
      v8f a = {};
      #pragma unroll
      for (int ks = 0; ks < 2; ++ks) {
        const u16* kp = Kh + (size_t)(kb + ntl * 16 + ln) * HD + ks * 32;
        v16bf bfr = make_frag(kp + 8 * h16, kp + 16 + 8 * h16);
        a = __builtin_amdgcn_wmma_f32_16x16x32_bf16(
                false, qfrag[ks], false, bfr, (short)0, a, false, false);
      }
      sc[ntl] = a;
    }

    // ---- multiplicative mask + 1/sqrt(E) scale ----
    const float inv32 = 0.03125f;
    #pragma unroll
    for (int ntl = 0; ntl < 8; ++ntl) {
      int nk = kb + ntl * 16 + ln;                // key index (C-layout N)
      #pragma unroll
      for (int r = 0; r < 8; ++r) {
        int mq = q0 + r + 8 * h16;                // query index (C-layout M)
        float s = sc[ntl][r];
        sc[ntl][r] = (nk <= mq) ? s * inv32 : 0.0f;
      }
    }

    // ---- streaming softmax: rowmax across lanes-within-half ----
    float alpha[8];
    #pragma unroll
    for (int r = 0; r < 8; ++r) {
      float m = sc[0][r];
      #pragma unroll
      for (int ntl = 1; ntl < 8; ++ntl) m = fmaxf(m, sc[ntl][r]);
      #pragma unroll
      for (int off = 1; off < 16; off <<= 1) m = fmaxf(m, __shfl_xor(m, off, 32));
      float mn = fmaxf(mrow[r], m);
      alpha[r] = __expf(mrow[r] - mn);
      mrow[r] = mn;
    }

    // ---- exp, rowsum, stash P into per-wave LDS (C-layout -> memory) ----
    float rsum[8];
    #pragma unroll
    for (int r = 0; r < 8; ++r) rsum[r] = 0.0f;
    #pragma unroll
    for (int ntl = 0; ntl < 8; ++ntl)
      #pragma unroll
      for (int r = 0; r < 8; ++r) {
        float p = __expf(sc[ntl][r] - mrow[r]);
        rsum[r] += p;
        pbuf[(r + 8 * h16) * P_STR + ntl * 16 + ln] = f2bf(p);
      }
    #pragma unroll
    for (int r = 0; r < 8; ++r) {
      float s = rsum[r];
      #pragma unroll
      for (int off = 1; off < 16; off <<= 1) s += __shfl_xor(s, off, 32);
      lrow[r] = lrow[r] * alpha[r] + s;
    }
    #pragma unroll
    for (int dt = 0; dt < 4; ++dt)
      #pragma unroll
      for (int r = 0; r < 8; ++r) oacc[dt][r] *= alpha[r];

    // ---- P @ V: A-fragments from LDS, B-fragments from transposed V ----
    // (per-wave LDS buffer: same-wave DS ops are in-order, no barrier needed)
    #pragma unroll
    for (int ls = 0; ls < 4; ++ls) {
      const u16* pr = pbuf + ln * P_STR + ls * 32;
      v16bf pfr = make_frag(pr + 8 * h16, pr + 16 + 8 * h16);
      #pragma unroll
      for (int dt = 0; dt < 4; ++dt) {
        const u16* vp = Vh + (size_t)(dt * 16 + ln) * S_LEN + kb + ls * 32;
        v16bf vfr = make_frag(vp + 8 * h16, vp + 16 + 8 * h16);
        oacc[dt] = __builtin_amdgcn_wmma_f32_16x16x32_bf16(
                       false, pfr, false, vfr, (short)0, oacc[dt], false, false);
      }
    }
  }

  // ---- normalize and store fp32 output [B][S][H*D] ----
  const int b = bh >> 4, h = bh & 15;
  #pragma unroll
  for (int r = 0; r < 8; ++r) {
    float invl = 1.0f / lrow[r];
    int s = q0 + r + 8 * h16;
    float* orow = Out + ((size_t)b * S_LEN + s) * EMB + h * HD;
    #pragma unroll
    for (int dt = 0; dt < 4; ++dt)
      orow[dt * 16 + ln] = oacc[dt][r] * invl;
  }
}

// ---------------------------------------------------------------------------
extern "C" void kernel_launch(void* const* d_in, const int* in_sizes, int n_in,
                              void* d_out, int out_size, void* d_ws, size_t ws_size,
                              hipStream_t stream) {
  const float* x  = (const float*)d_in[0];
  const float* Wq = (const float*)d_in[1];
  const float* Wk = (const float*)d_in[2];
  const float* Wv = (const float*)d_in[3];
  // d_in[4] (mask) unused: tril predicate computed analytically in-kernel.

  const size_t nX = (size_t)BATCH * S_LEN * EMB;   // 4M elems
  const size_t nW = (size_t)EMB * EMB;             // 1M elems
  const size_t nH = (size_t)BATCH * HEADS * S_LEN * HD; // 4M elems

  u16* Xbf = (u16*)d_ws;        // 8 MB
  u16* Wqb = Xbf + nX;          // 2 MB
  u16* Wkb = Wqb + nW;
  u16* Wvb = Wkb + nW;
  u16* Qb  = Wvb + nW;          // 8 MB each
  u16* Kb  = Qb + nH;
  u16* Vt  = Kb + nH;           // total 38 MB

  f32_to_bf16_kernel<<<dim3((int)(nX / 4 / 256)), 256, 0, stream>>>(x,  Xbf, (int)(nX / 4));
  f32_to_bf16_kernel<<<dim3((int)(nW / 4 / 256)), 256, 0, stream>>>(Wq, Wqb, (int)(nW / 4));
  f32_to_bf16_kernel<<<dim3((int)(nW / 4 / 256)), 256, 0, stream>>>(Wk, Wkb, (int)(nW / 4));
  f32_to_bf16_kernel<<<dim3((int)(nW / 4 / 256)), 256, 0, stream>>>(Wv, Wvb, (int)(nW / 4));

  dim3 g1(EMB / 128, (BATCH * S_LEN) / 128);
  qkv_gemm_kernel<0><<<g1, 256, 0, stream>>>(Xbf, Wqb, Qb);
  qkv_gemm_kernel<0><<<g1, 256, 0, stream>>>(Xbf, Wkb, Kb);
  qkv_gemm_kernel<1><<<g1, 256, 0, stream>>>(Xbf, Wvb, Vt);

  dim3 g2(S_LEN / 128, BATCH * HEADS);
  attn_kernel<<<g2, 256, 0, stream>>>(Qb, Kb, Vt, (float*)d_out);
}
